// MyMultiHeadAttention_8546984919358
// MI455X (gfx1250) — matrix operand order repaired and checked by reading, not compiled, not run on hardware
//
#include <hip/hip_runtime.h>
#include <hip/hip_bf16.h>

typedef _Float16 half_t;
typedef __attribute__((ext_vector_type(16))) _Float16 v16h;
typedef __attribute__((ext_vector_type(8)))  _Float16 v8h;
typedef __attribute__((ext_vector_type(8)))  float    v8f;

#define D_MODEL 1024
#define NH      16
#define DKK     64
#define SEQ     2048
#define NB      2
#define NROWS   (NB * SEQ)   // 4096

// ------------------------------------------------------------------
// CDNA5 async global->LDS copy (ASYNCcnt-tracked, ISA 15.18.3 op 98).
// LDS byte address = low 32 bits of the generic pointer (LDS aperture
// lives in addr[63:32]; addr[31:0] is the wave-relative LDS address).
// ------------------------------------------------------------------
__device__ __forceinline__ void cp_async16(half_t* lds_dst, const half_t* gsrc) {
  unsigned l = (unsigned)(size_t)lds_dst;
  asm volatile("global_load_async_to_lds_b128 %0, %1, off"
               :: "v"(l), "v"(gsrc)
               : "memory");
}
__device__ __forceinline__ void wait_async0() {
  asm volatile("s_wait_asynccnt 0" ::: "memory");
}

// ------------------------------------------------------------------
// WMMA fragment helpers (CDNA5 16x16x32 f16, layouts per ISA 7.12.2)
// ------------------------------------------------------------------
// A-matrix (16x32, MxK): lane m = lane&15, hi = lane>>4.
//   halves 0..7  -> k = hi*8 + 0..7        (contiguous 16B)
//   halves 8..15 -> k = 16 + hi*8 + 0..7   (contiguous 16B)
__device__ __forceinline__ v16h frag_a_ld(const half_t* base, int ld) {
  const int lane = threadIdx.x & 31;
  const int m  = lane & 15;
  const int hi = lane >> 4;
  const half_t* p = base + m * ld + hi * 8;
  v8h lo = *(const v8h*)(p);
  v8h hh = *(const v8h*)(p + 16);
  return __builtin_shufflevector(lo, hh, 0,1,2,3,4,5,6,7,8,9,10,11,12,13,14,15);
}

// B-matrix (32x16, KxN): lane n = lane&15, hi = lane>>4.
//   halves 0..7  -> k = hi*8 + 0..7        (contiguous 16B)
//   halves 8..15 -> k = 16 + hi*8 + 0..7   (contiguous 16B)
__device__ __forceinline__ v16h frag_b_ld(const half_t* base, int ld) {
  const int lane = threadIdx.x & 31;
  const int n  = lane & 15;
  const int hi = lane >> 4;
  const half_t* p = base + n * ld + hi * 8;
  v8h lo = *(const v8h*)(p);
  v8h hh = *(const v8h*)(p + 16);
  return __builtin_shufflevector(lo, hh, 0,1,2,3,4,5,6,7,8,9,10,11,12,13,14,15);
}

__device__ __forceinline__ v8f wmma_f16(v16h a, v16h b, v8f c) {
  return __builtin_amdgcn_wmma_f32_16x16x32_f16(false, a, false, b, (short)0, c,
                                                false, false);
}

__device__ __forceinline__ v8f vzero8() {
  v8f z = {0.f, 0.f, 0.f, 0.f, 0.f, 0.f, 0.f, 0.f};
  return z;
}

// C/D layout (16x16 f32): VGPR r, lanes 0-15 -> (M=r, N=lane),
// lanes 16-31 -> (M=r+8, N=lane-16). Row reductions over N are 16-lane
// shfl_xor butterflies.
__device__ __forceinline__ void reduce_max16(v8f& v) {
#pragma unroll
  for (int off = 1; off < 16; off <<= 1)
#pragma unroll
    for (int r = 0; r < 8; ++r)
      v[r] = fmaxf(v[r], __shfl_xor(v[r], off, 32));
}

__device__ __forceinline__ void reduce_sum16(v8f& v) {
#pragma unroll
  for (int off = 1; off < 16; off <<= 1)
#pragma unroll
    for (int r = 0; r < 8; ++r)
      v[r] += __shfl_xor(v[r], off, 32);
}

// ------------------------------------------------------------------
// fp32 -> f16 conversion
// ------------------------------------------------------------------
__global__ __launch_bounds__(256) void cvt_f32_f16(const float* __restrict__ in,
                                                   half_t* __restrict__ out, int n) {
  int i = blockIdx.x * 256 + threadIdx.x;
  if (i < n) out[i] = (half_t)in[i];
}

// ------------------------------------------------------------------
// GEMM: Y[M,N] = A[M,K] * W[N,K]^T  (einsum 'oi,bsi->bso')
// BM=BN=128, BK=32; 8 waves in 2x4 grid, wave tile 64x32
// (4x2 16x16 frags -> 8 WMMA per K-step). Double-buffered LDS fed by
// global_load_async_to_lds_b128, synced with s_wait_asynccnt.
// OUT_MODE 0: f16 scatter to [B,H,S,DK]; OUT_MODE 1: f32 row-major.
// ------------------------------------------------------------------
#define BM  128
#define BN  128
#define BK  32
#define LDA 40   // padded half-element stride (80B rows, 16B aligned)
#define LDB 40

__device__ __forceinline__ void gemm_stage_async(
    const half_t* __restrict__ A, const half_t* __restrict__ W,
    half_t* dA, half_t* dB, int gm0, int gn0, int K, int k0, int tid) {
#pragma unroll
  for (int c = tid; c < (BM * BK / 8); c += 256) {   // 512 16B chunks each
    int row = c >> 2;
    int col = (c & 3) * 8;
    cp_async16(&dA[row * LDA + col], &A[(size_t)(gm0 + row) * K + k0 + col]);
    cp_async16(&dB[row * LDB + col], &W[(size_t)(gn0 + row) * K + k0 + col]);
  }
}

template <int OUT_MODE>
__global__ __launch_bounds__(256) void gemm_wmma_kernel(
    const half_t* __restrict__ A, const half_t* __restrict__ W,
    half_t* __restrict__ outH, float* __restrict__ outF, int K) {
  __shared__ __align__(16) half_t sA[2][BM * LDA];
  __shared__ __align__(16) half_t sB[2][BN * LDB];

  const int tid  = threadIdx.x;
  const int gm0  = blockIdx.x * BM;
  const int gn0  = blockIdx.y * BN;
  const int wave = tid >> 5;
  const int wm   = wave >> 2;  // 0..1 -> 64-row band
  const int wn   = wave & 3;   // 0..3 -> 32-col band

  v8f acc[4][2];
#pragma unroll
  for (int i = 0; i < 4; ++i)
#pragma unroll
    for (int j = 0; j < 2; ++j) acc[i][j] = vzero8();

  const int nT = K / BK;
  gemm_stage_async(A, W, sA[0], sB[0], gm0, gn0, K, 0, tid);

  for (int kt = 0; kt < nT; ++kt) {
    wait_async0();       // this wave's async loads for buf[kt&1] are done
    __syncthreads();     // ...and every other wave's too
    if (kt + 1 < nT)
      gemm_stage_async(A, W, sA[(kt + 1) & 1], sB[(kt + 1) & 1],
                       gm0, gn0, K, (kt + 1) * BK, tid);

    const half_t* cA = sA[kt & 1];
    const half_t* cB = sB[kt & 1];
    v16h af[4], bf[2];
#pragma unroll
    for (int i = 0; i < 4; ++i)
      af[i] = frag_a_ld(cA + (wm * 64 + i * 16) * LDA, LDA);
#pragma unroll
    for (int j = 0; j < 2; ++j)
      bf[j] = frag_b_ld(cB + (wn * 32 + j * 16) * LDB, LDB);
#pragma unroll
    for (int i = 0; i < 4; ++i)
#pragma unroll
      for (int j = 0; j < 2; ++j)
        acc[i][j] = wmma_f16(af[i], bf[j], acc[i][j]);
  }

  const int lane = tid & 31;
  const int nl   = lane & 15;
  const int hiL  = lane >> 4;
#pragma unroll
  for (int mf = 0; mf < 4; ++mf)
#pragma unroll
    for (int nf = 0; nf < 2; ++nf)
#pragma unroll
      for (int r = 0; r < 8; ++r) {
        int gm  = gm0 + wm * 64 + mf * 16 + r + hiL * 8;
        int gn  = gn0 + wn * 32 + nf * 16 + nl;
        float v = acc[mf][nf][r];
        if (OUT_MODE == 0) {
          int b = gm >> 11, s = gm & (SEQ - 1);
          int h = gn >> 6,  d = gn & (DKK - 1);
          outH[(((size_t)(b * NH + h) * SEQ + s) << 6) + d] = (half_t)v;
        } else {
          outF[(size_t)gm * D_MODEL + gn] = v;
        }
      }
}

// ------------------------------------------------------------------
// RoPE (interleaved pairs) in-place on [B,H,S,DK] f16
// ------------------------------------------------------------------
__global__ __launch_bounds__(256) void rope_kernel(half_t* __restrict__ qk,
                                                   const int* __restrict__ tok,
                                                   int total) {
  int i = blockIdx.x * 256 + threadIdx.x;
  if (i >= total) return;
  int pr  = i & (DKK / 2 - 1);       // pair index 0..31
  int row = i >> 5;                  // (b*NH+h)*SEQ + s
  int s   = row & (SEQ - 1);
  size_t base = ((size_t)row << 6) + (pr << 1);
  float pos  = (float)tok[s];
  // theta^(-2k/dk) = exp(-k * ln(theta)/32)
  float ang = pos * __expf(-(float)pr * (9.210340371976184f / 32.0f));
  float c = __cosf(ang), sn = __sinf(ang);
  float x0 = (float)qk[base], x1 = (float)qk[base + 1];
  qk[base]     = (half_t)(x0 * c - x1 * sn);
  qk[base + 1] = (half_t)(x0 * sn + x1 * c);
}

// ------------------------------------------------------------------
// Flash attention: block = 128 q rows (8 waves x 16 rows), 64-key tiles,
// online softmax, causal. Q/K/V in [B,H,S,DK] f16; out [B,S,D] f16.
// ------------------------------------------------------------------
#define QT   128
#define KT   64
#define VLD  72   // padded LDS stride (144B rows, 16B aligned)

__global__ __launch_bounds__(256) void flash_attn_kernel(
    const half_t* __restrict__ Qh, const half_t* __restrict__ Kh,
    const half_t* __restrict__ Vh, half_t* __restrict__ Oh) {
  __shared__ __align__(16) half_t sVt[DKK * VLD];       // V^T tile [dk][key]
  __shared__ __align__(16) half_t sP[8 * 16 * VLD];     // per-wave P [16][VLD]

  const int tid  = threadIdx.x;
  const int lane = tid & 31;
  const int wave = tid >> 5;
  const int nl   = lane & 15;
  const int hiL  = lane >> 4;

  const int qt = blockIdx.x;             // q tile (0..15)
  const int h  = blockIdx.y;
  const int b  = blockIdx.z;
  const size_t baseBH = (size_t)(b * NH + h) * SEQ * DKK;

  const int qrow0 = qt * QT + wave * 16; // wave's first q row
  const int qlast = qrow0 + 15;

  // Q fragments (A operand, 16x64 split into two 16x32 frags), direct global.
  v16h qf[2];
#pragma unroll
  for (int kf = 0; kf < 2; ++kf)
    qf[kf] = frag_a_ld(Qh + baseBH + (size_t)qrow0 * DKK + kf * 32, DKK);

  v8f oacc[4];
#pragma unroll
  for (int i = 0; i < 4; ++i) oacc[i] = vzero8();
  v8f lrun = vzero8();
  v8f mrun;
#pragma unroll
  for (int r = 0; r < 8; ++r) mrun[r] = -__builtin_inff();

  const int ktiles = qt * 2 + 2;         // causal upper bound for this block

  for (int kt = 0; kt < ktiles; ++kt) {
    const int key0 = kt * KT;

    __syncthreads();
    // Stage V^T: read V[key][dk] coalesced, scatter into sVt[dk][key].
#pragma unroll
    for (int c = tid; c < (KT * DKK / 8); c += 256) {
      int key = c >> 3;
      int d0  = (c & 7) * 8;
      v8h vv = *(const v8h*)(Vh + baseBH + (size_t)(key0 + key) * DKK + d0);
#pragma unroll
      for (int j = 0; j < 8; ++j) sVt[(d0 + j) * VLD + key] = vv[j];
    }
    __syncthreads();

    if (key0 <= qlast) {
      // S = Q * K^T : K fragments straight from global (two 16B loads/lane, 32B apart)
      v8f sacc[4];
#pragma unroll
      for (int nf = 0; nf < 4; ++nf) {
        sacc[nf] = vzero8();
        const half_t* kb = Kh + baseBH + (size_t)(key0 + nf * 16) * DKK;
#pragma unroll
        for (int kf = 0; kf < 2; ++kf)
          sacc[nf] = wmma_f16(qf[kf], frag_b_ld(kb + kf * 32, DKK), sacc[nf]);
      }
      // scale (1/sqrt(64)) + causal mask in C layout
#pragma unroll
      for (int nf = 0; nf < 4; ++nf)
#pragma unroll
        for (int r = 0; r < 8; ++r) {
          int key = key0 + nf * 16 + nl;
          int q   = qrow0 + r + hiL * 8;
          float sv = sacc[nf][r] * 0.125f;
          sacc[nf][r] = (key <= q) ? sv : -__builtin_inff();
        }
      // online softmax
      v8f tmax;
#pragma unroll
      for (int r = 0; r < 8; ++r) tmax[r] = sacc[0][r];
#pragma unroll
      for (int nf = 1; nf < 4; ++nf)
#pragma unroll
        for (int r = 0; r < 8; ++r) tmax[r] = fmaxf(tmax[r], sacc[nf][r]);
      reduce_max16(tmax);

      v8f nmax, alpha;
#pragma unroll
      for (int r = 0; r < 8; ++r) {
        nmax[r]  = fmaxf(mrun[r], tmax[r]);
        alpha[r] = __expf(mrun[r] - nmax[r]);
      }
#pragma unroll
      for (int nf = 0; nf < 4; ++nf)
#pragma unroll
        for (int r = 0; r < 8; ++r) oacc[nf][r] *= alpha[r];

      half_t* Pw = &sP[wave * 16 * VLD];
      v8f tsum = vzero8();
#pragma unroll
      for (int nf = 0; nf < 4; ++nf)
#pragma unroll
        for (int r = 0; r < 8; ++r) {
          float pv = __expf(sacc[nf][r] - nmax[r]);
          tsum[r] += pv;
          Pw[(r + hiL * 8) * VLD + nf * 16 + nl] = (half_t)pv;
        }
      reduce_sum16(tsum);
#pragma unroll
      for (int r = 0; r < 8; ++r) {
        lrun[r] = lrun[r] * alpha[r] + tsum[r];
        mrun[r] = nmax[r];
      }

      // Wave-private LDS transpose: DS ops are in-order per wave; wait for
      // the scattered stores before cross-lane fragment reads.
      asm volatile("s_wait_dscnt 0" ::: "memory");

      v16h pf[2];
      pf[0] = frag_a_ld(Pw, VLD);
      pf[1] = frag_a_ld(Pw + 32, VLD);
      // O += P * V
#pragma unroll
      for (int nf = 0; nf < 4; ++nf)
#pragma unroll
        for (int kf = 0; kf < 2; ++kf)
          oacc[nf] = wmma_f16(
              pf[kf], frag_b_ld(&sVt[(nf * 16) * VLD + kf * 32], VLD), oacc[nf]);
    }
  }

  // normalize + write [B,S,D] f16 for the output projection
#pragma unroll
  for (int nf = 0; nf < 4; ++nf)
#pragma unroll
    for (int r = 0; r < 8; ++r) {
      float ov = oacc[nf][r] / lrun[r];
      int s = qrow0 + r + hiL * 8;
      int d = nf * 16 + nl;
      Oh[(size_t)(b * SEQ + s) * D_MODEL + h * DKK + d] = (half_t)ov;
    }
}

// ------------------------------------------------------------------
// Host launcher
// ------------------------------------------------------------------
extern "C" void kernel_launch(void* const* d_in, const int* in_sizes, int n_in,
                              void* d_out, int out_size, void* d_ws, size_t ws_size,
                              hipStream_t stream) {
  const float* x  = (const float*)d_in[0];
  const int*   tk = (const int*)d_in[1];
  const float* Wq = (const float*)d_in[2];
  const float* Wk = (const float*)d_in[3];
  const float* Wv = (const float*)d_in[4];
  const float* Wo = (const float*)d_in[5];
  float* out = (float*)d_out;

  char* ws = (char*)d_ws;
  size_t off = 0;
  const size_t nX = (size_t)NROWS * D_MODEL;   // 4,194,304
  const size_t nW = (size_t)D_MODEL * D_MODEL; // 1,048,576
  half_t* xh  = (half_t*)(ws + off); off += nX * 2;
  half_t* wqh = (half_t*)(ws + off); off += nW * 2;
  half_t* wkh = (half_t*)(ws + off); off += nW * 2;
  half_t* wvh = (half_t*)(ws + off); off += nW * 2;
  half_t* woh = (half_t*)(ws + off); off += nW * 2;
  half_t* qh  = (half_t*)(ws + off); off += nX * 2;
  half_t* kh  = (half_t*)(ws + off); off += nX * 2;
  half_t* vh  = (half_t*)(ws + off); off += nX * 2;
  half_t* ah  = (half_t*)(ws + off); off += nX * 2;

  cvt_f32_f16<<<(int)((nX + 255) / 256), 256, 0, stream>>>(x,  xh,  (int)nX);
  cvt_f32_f16<<<(int)((nW + 255) / 256), 256, 0, stream>>>(Wq, wqh, (int)nW);
  cvt_f32_f16<<<(int)((nW + 255) / 256), 256, 0, stream>>>(Wk, wkh, (int)nW);
  cvt_f32_f16<<<(int)((nW + 255) / 256), 256, 0, stream>>>(Wv, wvh, (int)nW);
  cvt_f32_f16<<<(int)((nW + 255) / 256), 256, 0, stream>>>(Wo, woh, (int)nW);

  dim3 gg(NROWS / BM, D_MODEL / BN);  // 32 x 8
  gemm_wmma_kernel<0><<<gg, 256, 0, stream>>>(xh, wqh, qh, nullptr, D_MODEL);
  gemm_wmma_kernel<0><<<gg, 256, 0, stream>>>(xh, wkh, kh, nullptr, D_MODEL);
  gemm_wmma_kernel<0><<<gg, 256, 0, stream>>>(xh, wvh, vh, nullptr, D_MODEL);

  int nR = NB * NH * SEQ * (DKK / 2);  // 2,097,152 rotation pairs
  rope_kernel<<<(nR + 255) / 256, 256, 0, stream>>>(qh, tk, nR);
  rope_kernel<<<(nR + 255) / 256, 256, 0, stream>>>(kh, tk, nR);

  dim3 fg(SEQ / QT, NH, NB);  // 16 x 16 x 2
  flash_attn_kernel<<<fg, 256, 0, stream>>>(qh, kh, vh, ah);

  gemm_wmma_kernel<1><<<gg, 256, 0, stream>>>(ah, woh, nullptr, out, D_MODEL);
}
